// SwinTransformerBlock_78494822302233
// MI455X (gfx1250) — matrix operand order repaired and checked by reading, not compiled, not run on hardware
//
#include <hip/hip_runtime.h>
#include <hip/hip_bf16.h>
#include <cstdint>

// ---------------------------------------------------------------------------
// Swin block for MI455X (gfx1250, wave32). All matmuls in bf16 WMMA
// (V_WMMA_F32_16X16X32_BF16), fp32 accumulate. GEMM mainloop uses CDNA5
// async global->LDS copies (ASYNCcnt) with double-buffered LDS tiles.
// ---------------------------------------------------------------------------

#define C_DIM   512
#define N_HEADS 16
#define HD      32
#define WSZ     8
#define SSZ     4
#define NTOK    64                   // tokens per window
#define NWIN    2048                 // 32 batches * 64 windows
#define MTOT    (NWIN * NTOK)        // 131072 token rows
#define HRES    64

typedef __attribute__((ext_vector_type(16))) __bf16 v16bf;
typedef __attribute__((ext_vector_type(8)))  float  v8f;

__device__ __forceinline__ unsigned short f32_bf16(float f) {
    unsigned u = __float_as_uint(f);
    u += 0x7FFFu + ((u >> 16) & 1u);          // round to nearest even
    return (unsigned short)(u >> 16);
}
__device__ __forceinline__ float bf16_f32(unsigned short h) {
    return __uint_as_float(((unsigned)h) << 16);
}

union FragU { uint4 u[2]; v16bf v; };

// A-matrix fragment (16x32 MxK, bf16), per ISA 7.12.2:
// lane: m = lane&15, kb = (lane&16)?8:0; elem i: k = kb + (i&7) + 16*(i>=8)
// -> two contiguous 8-elem (16B) chunks at k = kb and k = kb+16.
__device__ __forceinline__ v16bf frag_a(const unsigned short* base, int row0, int ld, int k0) {
    int lane = threadIdx.x & 31;
    int m  = lane & 15;
    int kb = (lane & 16) ? 8 : 0;
    const unsigned short* p = base + (size_t)(row0 + m) * ld + k0 + kb;
    FragU r;
    r.u[0] = *(const uint4*)(p);
    r.u[1] = *(const uint4*)(p + 16);
    return r.v;
}

// B-matrix fragment (32x16 KxN, bf16): lane n = lane&15, kb = (lane&16)?8:0;
// elem i: k = kb + (i&7) + 16*(i>=8)  -> two 8-elem runs of B[k][n] at fixed n.
// base is "N-major" (rows = output column n, ld along K) == weight layout W[N,K].
__device__ __forceinline__ v16bf frag_b(const unsigned short* base, int col0, int ld, int k0) {
    int lane = threadIdx.x & 31;
    int n  = lane & 15;
    int kb = (lane & 16) ? 8 : 0;
    const unsigned short* p = base + (size_t)(col0 + n) * ld + k0 + kb;
    FragU r;
    r.u[0] = *(const uint4*)(p);
    r.u[1] = *(const uint4*)(p + 16);
    return r.v;
}

__device__ __forceinline__ v8f wmma_bf16(v16bf a, v16bf b, v8f c) {
    return __builtin_amdgcn_wmma_f32_16x16x32_bf16(false, a, false, b, (short)0, c, false, false);
}

// CDNA5 async global->LDS 16B copy (GV mode): VDST = per-lane LDS byte address.
// Generic pointers to LDS carry the LDS offset in bits [31:0] (aperture encoding),
// so truncating a __shared__ generic pointer yields the required VDST value.
__device__ __forceinline__ void async_copy_b128(void* lds_dst, const void* gsrc) {
    unsigned lds_addr = (unsigned)(uintptr_t)lds_dst;
    asm volatile("global_load_async_to_lds_b128 %0, %1, off"
                 :: "v"(lds_addr), "v"(gsrc) : "memory");
}
__device__ __forceinline__ void wait_async_le3() {
    asm volatile("s_wait_asynccnt 3" ::: "memory");
}
__device__ __forceinline__ void wait_async_0() {
    asm volatile("s_wait_asynccnt 0" ::: "memory");
}

// ---------------------------------------------------------------------------
// fp32 -> bf16 weight conversion
// ---------------------------------------------------------------------------
__global__ void cvt_bf16_kernel(const float* __restrict__ src,
                                unsigned short* __restrict__ dst, int n) {
    int i = blockIdx.x * blockDim.x + threadIdx.x;
    if (i < n) dst[i] = f32_bf16(src[i]);
}

// ---------------------------------------------------------------------------
// LN1 + cyclic shift (-SSZ) + window partition, fp32 -> bf16
// ---------------------------------------------------------------------------
__global__ __launch_bounds__(256)
void ln1_shift_window(const float* __restrict__ x, const float* __restrict__ g,
                      const float* __restrict__ b, unsigned short* __restrict__ out) {
    int blk = blockIdx.x;            // windowed token id
    int wid = blk >> 6;              // window id
    int t   = blk & 63;
    int bat = wid >> 6;
    int win = wid & 63;
    int wh = win >> 3, ww = win & 7;
    int ti = t >> 3,   tj = t & 7;
    int sh = (wh * WSZ + ti + SSZ) & (HRES - 1);   // roll(x,-SSZ): out[i]=in[i+SSZ]
    int sw = (ww * WSZ + tj + SSZ) & (HRES - 1);
    const float* row = x + ((size_t)bat * HRES * HRES + sh * HRES + sw) * C_DIM;

    int tid = threadIdx.x;
    float v0 = row[tid], v1 = row[tid + 256];
    float s = v0 + v1, q = v0 * v0 + v1 * v1;
    for (int m = 16; m; m >>= 1) { s += __shfl_xor(s, m, 32); q += __shfl_xor(q, m, 32); }
    __shared__ float rs[8], rq[8];
    if ((tid & 31) == 0) { rs[tid >> 5] = s; rq[tid >> 5] = q; }
    __syncthreads();
    float S = 0.f, Q = 0.f;
    for (int i = 0; i < 8; ++i) { S += rs[i]; Q += rq[i]; }
    float mu  = S * (1.f / C_DIM);
    float inv = rsqrtf(Q * (1.f / C_DIM) - mu * mu + 1e-5f);
    unsigned short* orow = out + (size_t)blk * C_DIM;
    orow[tid]       = f32_bf16((v0 - mu) * inv * g[tid]       + b[tid]);
    orow[tid + 256] = f32_bf16((v1 - mu) * inv * g[tid + 256] + b[tid + 256]);
}

// ---------------------------------------------------------------------------
// Generic WMMA GEMM:  out[M,N] = A[M,K](bf16) @ W[N,K]^T(bf16) + bias
// block = 256 thr (8 waves), tile 64M x 128N, K-step 32.
// Mainloop: double-buffered LDS tiles fed by async global->LDS (3 x b128 per
// thread per tile); s_wait_asynccnt 3 overlaps tile k+1 fetch with tile k WMMAs.
// EPI 0: store bf16 ; EPI 1: exact GELU, store bf16 ; EPI 2: += resid, store fp32
// ---------------------------------------------------------------------------
template <int EPI>
__global__ __launch_bounds__(256)
void gemm_wmma(const unsigned short* __restrict__ A, const unsigned short* __restrict__ W,
               const float* __restrict__ bias, unsigned short* __restrict__ outb,
               float* __restrict__ outf, const float* __restrict__ resid,
               int N, int K) {
    __shared__ __align__(16) unsigned short sA[2][64 * 32];
    __shared__ __align__(16) unsigned short sB[2][128 * 32];

    int tid  = threadIdx.x;
    int wave = tid >> 5, lane = tid & 31;
    int wm = wave & 3, wn = wave >> 2;           // 4 waves along M, 2 along N
    int bm = blockIdx.x * 64, bn = blockIdx.y * 128;

    // per-thread async-copy slots: A tile 64x32 = 256 x b128, B tile 128x32 = 512 x b128
    int ar = tid >> 2, ac8 = (tid & 3) * 8;
    int br0 = tid >> 2,         bc0 = (tid & 3) * 8;
    int br1 = (tid + 256) >> 2, bc1 = (tid & 3) * 8;

    auto issue_tile = [&](int buf, int k0) {
        async_copy_b128(&sA[buf][ar * 32 + ac8],  &A[(size_t)(bm + ar) * K + k0 + ac8]);
        async_copy_b128(&sB[buf][br0 * 32 + bc0], &W[(size_t)(bn + br0) * K + k0 + bc0]);
        async_copy_b128(&sB[buf][br1 * 32 + bc1], &W[(size_t)(bn + br1) * K + k0 + bc1]);
    };

    v8f acc[4];
    for (int nt = 0; nt < 4; ++nt)
        for (int e = 0; e < 8; ++e) acc[nt][e] = 0.f;

    issue_tile(0, 0);
    int cur = 0;
    for (int k0 = 0; k0 < K; k0 += 32) {
        int nxt = cur ^ 1;
        bool more = (k0 + 32) < K;
        if (more) { issue_tile(nxt, k0 + 32); wait_async_le3(); }
        else      { wait_async_0(); }
        __syncthreads();                          // all waves' tile(cur) landed in LDS

        v16bf af = frag_a(sA[cur], wm * 16, 32, 0);
        for (int nt = 0; nt < 4; ++nt) {
            v16bf bf = frag_b(sB[cur], wn * 64 + nt * 16, 32, 0);
            acc[nt] = wmma_bf16(af, bf, acc[nt]);
        }
        __syncthreads();                          // done reading tile(cur) before reuse
        cur = nxt;
    }

    // C layout: lanes 0-15 -> M = v, lanes 16-31 -> M = v+8 ; N = lane&15
    int mbase = bm + wm * 16 + ((lane & 16) ? 8 : 0);
    int nbase = bn + wn * 64 + (lane & 15);
    for (int nt = 0; nt < 4; ++nt) {
        int n = nbase + nt * 16;
        float bv = bias[n];
        for (int v = 0; v < 8; ++v) {
            int m = mbase + v;
            float val = acc[nt][v] + bv;
            if (EPI == 1) val = 0.5f * val * (1.f + erff(val * 0.70710678118654752f));
            size_t o = (size_t)m * N + n;
            if (EPI == 2) outf[o] = resid[o] + val;
            else          outb[o] = f32_bf16(val);
        }
    }
}

// ---------------------------------------------------------------------------
// Windowed attention: 1 WG (128 thr = 4 waves) per (window, head).
// S = q k^T (WMMA), +rel-pos bias +shift mask (closed-form), softmax,
// P -> LDS (re-swizzle C->A layout), out = P v (WMMA).
// ---------------------------------------------------------------------------
__global__ __launch_bounds__(128)
void attn_window(const unsigned short* __restrict__ qkv, const float* __restrict__ rpb,
                 unsigned short* __restrict__ out) {
    int wid  = blockIdx.x;           // 0..2047
    int head = blockIdx.y;           // 0..15
    __shared__ __align__(16) unsigned short sq[NTOK * HD];    // q row-major [64][32]
    __shared__ __align__(16) unsigned short sk[NTOK * HD];    // k row-major [64][32]
    __shared__ __align__(16) unsigned short svT[HD * NTOK];   // v transposed [32][64]
    __shared__ __align__(16) unsigned short sp[NTOK * NTOK];  // probs [64][64]

    int tid = threadIdx.x, wave = tid >> 5, lane = tid & 31;
    const unsigned short* base = qkv + (size_t)wid * NTOK * (3 * C_DIM) + head * HD;

    // q & k via async global->LDS (2 x b128 per thread each)
    for (int it = 0; it < 2; ++it) {
        int idx = tid + it * 128;
        int r = idx >> 2, c8 = (idx & 3) * 8;
        async_copy_b128(&sq[r * HD + c8], &base[(size_t)r * (3 * C_DIM) + c8]);
        async_copy_b128(&sk[r * HD + c8], &base[(size_t)r * (3 * C_DIM) + C_DIM + c8]);
    }
    for (int idx = tid; idx < NTOK * HD; idx += 128) {   // v, transposed store
        int r = idx >> 5, d = idx & 31;
        svT[d * NTOK + r] = base[(size_t)r * (3 * C_DIM) + 2 * C_DIM + d];
    }
    wait_async_0();
    __syncthreads();

    // ---- S = q @ k^T : A = q rows [16*wave, +16), B = k rows (as N-major), K=32
    v16bf qf = frag_a(sq, wave * 16, HD, 0);
    v8f s[4];
    for (int nt = 0; nt < 4; ++nt) {
        v8f z; for (int e = 0; e < 8; ++e) z[e] = 0.f;
        s[nt] = wmma_bf16(qf, frag_b(sk, nt * 16, HD, 0), z);
    }

    const float scale = 0.17677669529663687f;    // 1/sqrt(32)
    int lh = (lane >> 4) & 1, nc = lane & 15;
    int win = wid & 63, wh = win >> 3, ww = win & 7;

    float logit[4][8];
    for (int nt = 0; nt < 4; ++nt)
        for (int v = 0; v < 8; ++v) {
            int m = wave * 16 + v + 8 * lh;
            int n = nt * 16 + nc;
            int mi = m >> 3, mj = m & 7, ni = n >> 3, nj = n & 7;
            int ridx = (mi - ni + WSZ - 1) * (2 * WSZ - 1) + (mj - nj + WSZ - 1);
            float bv = rpb[ridx * N_HEADS + head];
            int hm = wh * WSZ + mi, wm_ = ww * WSZ + mj;
            int hn = wh * WSZ + ni, wn_ = ww * WSZ + nj;
            int zm = ((hm < 56) ? 0 : (hm < 60) ? 1 : 2) * 3 + ((wm_ < 56) ? 0 : (wm_ < 60) ? 1 : 2);
            int zn = ((hn < 56) ? 0 : (hn < 60) ? 1 : 2) * 3 + ((wn_ < 56) ? 0 : (wn_ < 60) ? 1 : 2);
            logit[nt][v] = s[nt][v] * scale + bv + ((zm != zn) ? -100.f : 0.f);
        }

    // ---- row softmax (row lives in one 16-lane half across 4 N-tiles)
    float rmax[8], rsum[8];
    for (int v = 0; v < 8; ++v) {
        float mx = fmaxf(fmaxf(logit[0][v], logit[1][v]), fmaxf(logit[2][v], logit[3][v]));
        for (int msk = 8; msk; msk >>= 1) mx = fmaxf(mx, __shfl_xor(mx, msk, 32));
        rmax[v] = mx; rsum[v] = 0.f;
    }
    for (int nt = 0; nt < 4; ++nt)
        for (int v = 0; v < 8; ++v) {
            logit[nt][v] = __expf(logit[nt][v] - rmax[v]);
            rsum[v] += logit[nt][v];
        }
    for (int v = 0; v < 8; ++v) {
        float su = rsum[v];
        for (int msk = 8; msk; msk >>= 1) su += __shfl_xor(su, msk, 32);
        rsum[v] = 1.f / su;
    }
    for (int nt = 0; nt < 4; ++nt)
        for (int v = 0; v < 8; ++v) {
            int m = wave * 16 + v + 8 * lh;
            sp[m * NTOK + nt * 16 + nc] = f32_bf16(logit[nt][v] * rsum[v]);
        }
    __syncthreads();

    // ---- out = P @ V : K = 64 (2 steps), N = 32 (2 tiles), B = v^T (N-major)
    v8f o[2];
    for (int nt = 0; nt < 2; ++nt)
        for (int e = 0; e < 8; ++e) o[nt][e] = 0.f;
    for (int ks = 0; ks < 2; ++ks) {
        v16bf pf = frag_a(sp, wave * 16, NTOK, ks * 32);
        for (int nt = 0; nt < 2; ++nt)
            o[nt] = wmma_bf16(pf, frag_b(svT, nt * 16, NTOK, ks * 32), o[nt]);
    }
    for (int nt = 0; nt < 2; ++nt)
        for (int v = 0; v < 8; ++v) {
            int m = wave * 16 + v + 8 * lh;
            int d = nt * 16 + nc;
            out[(size_t)(wid * NTOK + m) * C_DIM + head * HD + d] = f32_bf16(o[nt][v]);
        }
}

// ---------------------------------------------------------------------------
// window reverse + unshift (+SSZ) + residual (fp32 -> d_out) + LN2 (-> bf16)
// ---------------------------------------------------------------------------
__global__ __launch_bounds__(256)
void unshift_res_ln2(const float* __restrict__ x, const unsigned short* __restrict__ pw,
                     const float* __restrict__ g, const float* __restrict__ b,
                     float* __restrict__ xres, unsigned short* __restrict__ lnout) {
    int blk = blockIdx.x;            // original token id
    int bat = blk >> 12;
    int hw = blk & 4095;
    int h = hw >> 6, w = hw & 63;
    int sh = (h + HRES - SSZ) & (HRES - 1);   // roll(+SSZ): out[i]=in[i-SSZ]
    int sw = (w + HRES - SSZ) & (HRES - 1);
    int wid = bat * 64 + (sh >> 3) * 8 + (sw >> 3);
    int t   = (sh & 7) * 8 + (sw & 7);
    const unsigned short* prow = pw + (size_t)(wid * NTOK + t) * C_DIM;
    const float* xrow = x + (size_t)blk * C_DIM;

    int tid = threadIdx.x;
    float v0 = xrow[tid]       + bf16_f32(prow[tid]);
    float v1 = xrow[tid + 256] + bf16_f32(prow[tid + 256]);
    float* orow = xres + (size_t)blk * C_DIM;
    orow[tid] = v0; orow[tid + 256] = v1;

    float s = v0 + v1, q = v0 * v0 + v1 * v1;
    for (int m = 16; m; m >>= 1) { s += __shfl_xor(s, m, 32); q += __shfl_xor(q, m, 32); }
    __shared__ float rs[8], rq[8];
    if ((tid & 31) == 0) { rs[tid >> 5] = s; rq[tid >> 5] = q; }
    __syncthreads();
    float S = 0.f, Q = 0.f;
    for (int i = 0; i < 8; ++i) { S += rs[i]; Q += rq[i]; }
    float mu  = S * (1.f / C_DIM);
    float inv = rsqrtf(Q * (1.f / C_DIM) - mu * mu + 1e-5f);
    unsigned short* lrow = lnout + (size_t)blk * C_DIM;
    lrow[tid]       = f32_bf16((v0 - mu) * inv * g[tid]       + b[tid]);
    lrow[tid + 256] = f32_bf16((v1 - mu) * inv * g[tid + 256] + b[tid + 256]);
}

// ---------------------------------------------------------------------------
extern "C" void kernel_launch(void* const* d_in, const int* in_sizes, int n_in,
                              void* d_out, int out_size, void* d_ws, size_t ws_size,
                              hipStream_t stream) {
    const float* x      = (const float*)d_in[0];
    const float* n1g    = (const float*)d_in[1];
    const float* n1b    = (const float*)d_in[2];
    const float* qkv_w  = (const float*)d_in[3];
    const float* qkv_b  = (const float*)d_in[4];
    const float* proj_w = (const float*)d_in[5];
    const float* proj_b = (const float*)d_in[6];
    const float* rpb    = (const float*)d_in[7];
    const float* n2g    = (const float*)d_in[8];
    const float* n2b    = (const float*)d_in[9];
    const float* fc1_w  = (const float*)d_in[10];
    const float* fc1_b  = (const float*)d_in[11];
    const float* fc2_w  = (const float*)d_in[12];
    const float* fc2_b  = (const float*)d_in[13];
    float* out = (float*)d_out;

    // workspace partition (bf16 elements); regions reused across phases
    unsigned short* wsb   = (unsigned short*)d_ws;
    unsigned short* wqkv  = wsb;                                   // 1536*512
    unsigned short* wproj = wqkv  + (size_t)1536 * 512;            // 512*512
    unsigned short* wfc1  = wproj + (size_t)512 * 512;             // 2048*512
    unsigned short* wfc2  = wfc1  + (size_t)2048 * 512;            // 512*2048
    unsigned short* R1    = wfc2  + (size_t)512 * 2048;            // hw -> proj_out   [M,512]
    unsigned short* R3    = R1 + (size_t)MTOT * C_DIM;             // attn_out -> ln2  [M,512]
    unsigned short* R2    = R3 + (size_t)MTOT * C_DIM;             // qkv -> fc1_out   [M,2048]

    // 1) weights -> bf16
    cvt_bf16_kernel<<<(1536 * 512 + 255) / 256, 256, 0, stream>>>(qkv_w,  wqkv,  1536 * 512);
    cvt_bf16_kernel<<<(512 * 512 + 255) / 256, 256, 0, stream>>>(proj_w, wproj, 512 * 512);
    cvt_bf16_kernel<<<(2048 * 512 + 255) / 256, 256, 0, stream>>>(fc1_w,  wfc1,  2048 * 512);
    cvt_bf16_kernel<<<(512 * 2048 + 255) / 256, 256, 0, stream>>>(fc2_w,  wfc2,  512 * 2048);

    // 2) LN1 + shift + window partition
    ln1_shift_window<<<MTOT, 256, 0, stream>>>(x, n1g, n1b, R1);

    // 3) QKV GEMM: [M,512] x [1536,512]^T -> bf16 [M,1536]
    gemm_wmma<0><<<dim3(MTOT / 64, 1536 / 128), 256, 0, stream>>>(R1, wqkv, qkv_b, R2, nullptr, nullptr, 1536, 512);

    // 4) attention per (window, head)
    attn_window<<<dim3(NWIN, N_HEADS), 128, 0, stream>>>(R2, rpb, R3);

    // 5) proj GEMM: [M,512] x [512,512]^T -> bf16 [M,512] (windowed order)
    gemm_wmma<0><<<dim3(MTOT / 64, 512 / 128), 256, 0, stream>>>(R3, wproj, proj_b, R1, nullptr, nullptr, 512, 512);

    // 6) window-reverse + residual (x_res -> d_out) + LN2 (-> R3)
    unshift_res_ln2<<<MTOT, 256, 0, stream>>>(x, R1, n2g, n2b, out, R3);

    // 7) FC1 GEMM + exact GELU -> bf16 [M,2048]
    gemm_wmma<1><<<dim3(MTOT / 64, 2048 / 128), 256, 0, stream>>>(R3, wfc1, fc1_b, R2, nullptr, nullptr, 2048, 512);

    // 8) FC2 GEMM + residual (reads x_res from d_out, writes final fp32)
    gemm_wmma<2><<<dim3(MTOT / 64, 512 / 128), 256, 0, stream>>>(R2, wfc2, fc2_b, nullptr, out, out, 512, 2048);
}
